// Time_Series_Multi_Head_Attention_55490977464586
// MI455X (gfx1250) — matrix operand order repaired and checked by reading, not compiled, not run on hardware
//
#include <hip/hip_runtime.h>

#define B_  8
#define L_  512
#define D_  1024
#define H_  16
#define DK_ 64

typedef __attribute__((ext_vector_type(16))) __bf16 v16bf;
typedef __attribute__((ext_vector_type(8)))  __bf16 v8bf;
typedef __attribute__((ext_vector_type(8)))  float  v8f;

// ---------- helpers ----------

__device__ __forceinline__ __bf16 f2bf(float f) {
    union { float f; unsigned u; } a; a.f = f;
    unsigned u = a.u + 0x7FFFu + ((a.u >> 16) & 1u);   // round-to-nearest-even
    union { unsigned short s; __bf16 b; } c; c.s = (unsigned short)(u >> 16);
    return c.b;
}

__device__ __forceinline__ v8f wmma_bf(v16bf a, v16bf b, v8f c) {
    // D = A(16x32 bf16) * B(32x16 bf16) + C(16x16 f32)
    return __builtin_amdgcn_wmma_f32_16x16x32_bf16(
        /*neg_a=*/false, a, /*neg_b=*/false, b,
        /*c_mod=*/(short)0, c, /*reuse_a=*/false, /*reuse_b=*/false);
}

// A fragment: 16x32 (MxK) bf16, row-major source, row stride lda (elements).
// Contiguous k order: lane l -> row l%16; g=l/16: halves 0..15 = K[16g..16g+15], K ascending
__device__ __forceinline__ v16bf load_a_frag(const __bf16* A, int lda, int lane, int k0) {
    int row = lane & 15, g = (lane >> 4) & 1;
    const __bf16* p = A + (size_t)row * lda + k0;
    return *(const v16bf*)(p + 16 * g);
}

// B fragment: 32x16 (KxN) bf16, NT form: WMMA-B[k,n] = Brm[n,k] (Brm row-major, ld ldb).
// Contiguous k order: lane l -> col l%16; lanes 0-15 hold K=0..15, lanes 16-31 K=16..31, K ascending.
__device__ __forceinline__ v16bf load_b_frag(const __bf16* Brm, int ldb, int lane, int k0) {
    int col = lane & 15, g = (lane >> 4) & 1;
    return *(const v16bf*)(Brm + (size_t)col * ldb + k0 + 16 * g);
}

__device__ __forceinline__ float block_red_max(float v, float* red) {
#pragma unroll
    for (int o = 16; o > 0; o >>= 1) v = fmaxf(v, __shfl_xor(v, o, 32));
    __syncthreads();
    if ((threadIdx.x & 31) == 0) red[threadIdx.x >> 5] = v;
    __syncthreads();
    float r = red[threadIdx.x & 7];
#pragma unroll
    for (int o = 4; o > 0; o >>= 1) r = fmaxf(r, __shfl_xor(r, o, 8));
    return r;
}

__device__ __forceinline__ float block_red_sum(float v, float* red) {
#pragma unroll
    for (int o = 16; o > 0; o >>= 1) v += __shfl_xor(v, o, 32);
    __syncthreads();
    if ((threadIdx.x & 31) == 0) red[threadIdx.x >> 5] = v;
    __syncthreads();
    float r = red[threadIdx.x & 7];
#pragma unroll
    for (int o = 4; o > 0; o >>= 1) r += __shfl_xor(r, o, 8);
    return r;
}

// ---------- kernels ----------

__global__ void __launch_bounds__(256)
cvt_f32_bf16(const float* __restrict__ in, __bf16* __restrict__ out, size_t n) {
    size_t i = (size_t)blockIdx.x * blockDim.x + threadIdx.x;
    if (i < n) out[i] = f2bf(in[i]);
}

// Register-blocked NT GEMM: C[m,n] = sum_k A[m,k]*Brm[n,k] (+bias[n]).
// One wave computes a 64x64 block (4x4 WMMA tiles, 16 f32 accumulators);
// 8 waves arranged 2(M) x 4(N) -> 128x256 block tile.
// mode: 0=bf16 row-major, 1=bf16 per-batch transposed ([b,n,l], row=b*L_+l), 2=f32 row-major.
__global__ void __launch_bounds__(256)
gemm_nt(const __bf16* __restrict__ A, int lda, size_t strideA,
        const __bf16* __restrict__ Bm, int ldb, size_t strideB,
        const float* __restrict__ bias, int K,
        float* __restrict__ outF, __bf16* __restrict__ outB, size_t strideO,
        int mode, int ldo) {
    int z = blockIdx.z;
    A  += (size_t)z * strideA;
    Bm += (size_t)z * strideB;
    int lane = threadIdx.x & 31, wave = threadIdx.x >> 5;
    int wm = wave >> 2;          // 0..1
    int wn = wave & 3;           // 0..3
    int m0 = blockIdx.x * 128 + wm * 64;
    int n0 = blockIdx.y * 256 + wn * 64;

    const __bf16* Ab[4];
    const __bf16* Bb[4];
#pragma unroll
    for (int i = 0; i < 4; i++) Ab[i] = A  + (size_t)(m0 + 16 * i) * lda;
#pragma unroll
    for (int j = 0; j < 4; j++) Bb[j] = Bm + (size_t)(n0 + 16 * j) * ldb;

    v8f acc[4][4] = {};
    for (int k = 0; k < K; k += 32) {
        // four K-blocks (128) ahead prefetch (lowers to global_prefetch_b8)
        if (k + 128 < K) {
            __builtin_prefetch(Ab[0] + (size_t)(lane & 15) * lda + k + 128, 0, 0);
            __builtin_prefetch(Bb[0] + (size_t)(lane & 15) * ldb + k + 128, 0, 0);
        }
        v16bf af[4], bf[4];
#pragma unroll
        for (int i = 0; i < 4; i++) af[i] = load_a_frag(Ab[i], lda, lane, k);
#pragma unroll
        for (int j = 0; j < 4; j++) bf[j] = load_b_frag(Bb[j], ldb, lane, k);
#pragma unroll
        for (int i = 0; i < 4; i++)
#pragma unroll
            for (int j = 0; j < 4; j++)
                acc[i][j] = wmma_bf(af[i], bf[j], acc[i][j]);
    }

    int g = (lane >> 4) & 1, r16 = lane & 15;
#pragma unroll
    for (int j = 0; j < 4; j++) {
        int col = n0 + 16 * j + r16;
        float bv = bias ? bias[col] : 0.0f;
#pragma unroll
        for (int i = 0; i < 4; i++) {
#pragma unroll
            for (int jj = 0; jj < 8; jj++) {
                int row = m0 + 16 * i + jj + 8 * g;
                float v = acc[i][j][jj] + bv;
                if (mode == 0) {
                    outB[(size_t)z * strideO + (size_t)row * ldo + col] = f2bf(v);
                } else if (mode == 1) {
                    int bb = row >> 9, l = row & (L_ - 1);
                    outB[((size_t)bb * ldo + col) * L_ + l] = f2bf(v);
                } else {
                    outF[(size_t)z * strideO + (size_t)row * ldo + col] = v;
                }
            }
        }
    }
}

// common[b,i,j] = sum_m lambdas[1+m]*softmax(-imf_m[b,i,:])[j]
//               + lambdas[6]*adj[b,i,j]/(rowsum(adj)+1e-6), masked keys -> -inf
__global__ void __launch_bounds__(256)
common_kernel(const float* __restrict__ adj,
              const float* __restrict__ i1, const float* __restrict__ i2,
              const float* __restrict__ i3, const float* __restrict__ i4,
              const float* __restrict__ i5, const float* __restrict__ lambdas,
              const int* __restrict__ mask, __bf16* __restrict__ common) {
    __shared__ float red[8];
    int row = blockIdx.x;                 // 0 .. B_*L_-1
    int b = row >> 9;
    size_t base = (size_t)row * L_;
    const int* mrow = mask + b * L_;
    int c0 = threadIdx.x, c1 = threadIdx.x + 256;
    int m0v = mrow[c0], m1v = mrow[c1];
    float acc0 = 0.f, acc1 = 0.f;
    const float* imfs[5] = { i1, i2, i3, i4, i5 };
    for (int m = 0; m < 5; m++) {
        const float* p = imfs[m] + base;
        float x0 = m0v ? -p[c0] : -__builtin_inff();
        float x1 = m1v ? -p[c1] : -__builtin_inff();
        float mx = block_red_max(fmaxf(x0, x1), red);
        float e0 = __expf(x0 - mx), e1 = __expf(x1 - mx);
        float s = block_red_sum(e0 + e1, red);
        float w = lambdas[1 + m] / s;
        acc0 += e0 * w; acc1 += e1 * w;
    }
    float a0 = adj[base + c0], a1 = adj[base + c1];
    float s = block_red_sum(a0 + a1, red);
    float w = lambdas[6] / (s + 1e-6f);
    acc0 += a0 * w; acc1 += a1 * w;
    common[base + c0] = f2bf(acc0);
    common[base + c1] = f2bf(acc1);
}

// Fused per-(b,h,16-row q-tile) attention:
//   S = (q kT)/8 -> mask -> softmax -> P(bf16) -> out = lambda0*(P v) + xc
__global__ void __launch_bounds__(128)
attn_kernel(const __bf16* __restrict__ qb, const __bf16* __restrict__ kb,
            const __bf16* __restrict__ vt, const float* __restrict__ xc,
            const float* __restrict__ lambdas, const int* __restrict__ mask,
            __bf16* __restrict__ attn) {
    __shared__ __align__(32) float  S[16 * L_];   // 32 KB
    __shared__ __align__(32) __bf16 P[16 * L_];   // 16 KB
    int qt = blockIdx.x, h = blockIdx.y, b = blockIdx.z;
    int lane = threadIdx.x & 31, wave = threadIdx.x >> 5;
    int g = (lane >> 4) & 1, r16 = lane & 15;

    const __bf16* qbase = qb + ((size_t)(b * L_) + qt * 16) * D_ + h * DK_;
    const __bf16* kbase = kb + (size_t)(b * L_) * D_ + h * DK_;
    v16bf a0 = load_a_frag(qbase, D_, lane, 0);    // K = 0..31 of dk
    v16bf a1 = load_a_frag(qbase, D_, lane, 32);   // K = 32..63

    // scores: waves stripe over the 32 key tiles
    for (int kt = wave; kt < L_ / 16; kt += 4) {
        const __bf16* kp = kbase + (size_t)(kt * 16) * D_;
        v16bf b0 = load_b_frag(kp, D_, lane, 0);
        v16bf b1 = load_b_frag(kp, D_, lane, 32);
        v8f acc = {};
        acc = wmma_bf(a0, b0, acc);
        acc = wmma_bf(a1, b1, acc);
#pragma unroll
        for (int j = 0; j < 8; j++)
            S[(j + 8 * g) * L_ + kt * 16 + r16] = acc[j] * 0.125f; // 1/sqrt(64)
    }
    __syncthreads();

    // row softmax: wave w owns rows w, w+4, w+8, w+12 (32 lanes x 16 cols each)
    const int* mrow = mask + b * L_;
    for (int r = wave; r < 16; r += 4) {
        float vals[16]; float mx = -1e30f;
#pragma unroll
        for (int i = 0; i < 16; i++) {
            int c = lane + 32 * i;
            float s = S[r * L_ + c];
            if (mrow[c] == 0) s = -1e12f;
            vals[i] = s; mx = fmaxf(mx, s);
        }
#pragma unroll
        for (int o = 16; o > 0; o >>= 1) mx = fmaxf(mx, __shfl_xor(mx, o, 32));
        float sum = 0.f;
#pragma unroll
        for (int i = 0; i < 16; i++) { vals[i] = __expf(vals[i] - mx); sum += vals[i]; }
#pragma unroll
        for (int o = 16; o > 0; o >>= 1) sum += __shfl_xor(sum, o, 32);
        float inv = 1.0f / sum;
#pragma unroll
        for (int i = 0; i < 16; i++) P[r * L_ + lane + 32 * i] = f2bf(vals[i] * inv);
    }
    __syncthreads();

    // out tile: wave w owns dk columns [16w, 16w+16); K loop over 512 in steps of 32
    int nt = wave;
    const __bf16* vbase = vt + (size_t)b * D_ * L_ + (size_t)(h * DK_ + nt * 16) * L_;
    v8f acc = {};
    for (int kt = 0; kt < L_ / 32; kt++) {
        v16bf a  = load_a_frag(P, L_, lane, kt * 32);
        v16bf bb = load_b_frag(vbase, L_, lane, kt * 32);
        acc = wmma_bf(a, bb, acc);
    }
    float l0 = lambdas[0];
    int col = h * DK_ + nt * 16 + r16;
#pragma unroll
    for (int j = 0; j < 8; j++) {
        int q = qt * 16 + j + 8 * g;
        size_t idx = ((size_t)(b * L_) + q) * D_ + col;
        attn[idx] = f2bf(l0 * acc[j] + xc[idx]);
    }
}

// ---------- launch ----------

extern "C" void kernel_launch(void* const* d_in, const int* in_sizes, int n_in,
                              void* d_out, int out_size, void* d_ws, size_t ws_size,
                              hipStream_t stream) {
    (void)in_sizes; (void)n_in; (void)out_size; (void)ws_size;
    const float* Query = (const float*)d_in[0];
    const float* Key   = (const float*)d_in[1];
    const float* Value = (const float*)d_in[2];
    const float* adj   = (const float*)d_in[3];
    const float* imf1  = (const float*)d_in[4];
    const float* imf2  = (const float*)d_in[5];
    const float* imf3  = (const float*)d_in[6];
    const float* imf4  = (const float*)d_in[7];
    const float* imf5  = (const float*)d_in[8];
    const float* Wq = (const float*)d_in[9];   const float* bq = (const float*)d_in[10];
    const float* Wk = (const float*)d_in[11];  const float* bk = (const float*)d_in[12];
    const float* Wv = (const float*)d_in[13];  const float* bv = (const float*)d_in[14];
    const float* Wo = (const float*)d_in[15];  const float* bo = (const float*)d_in[16];
    const float* lambdas = (const float*)d_in[17];
    const int*   mask    = (const int*)d_in[18];
    float* out = (float*)d_out;

    const size_t NE = (size_t)B_ * L_ * D_;   // 4,194,304
    const size_t WE = (size_t)D_ * D_;        // 1,048,576
    char* ws = (char*)d_ws;
    // region layout (bytes); xin region is reused by common+xc after projections
    __bf16* xq     = (__bf16*)(ws + 0);
    __bf16* xk     = (__bf16*)(ws + NE * 2);
    __bf16* xv     = (__bf16*)(ws + NE * 4);
    __bf16* common = (__bf16*)(ws + 0);          // B*L*L bf16 = NE bytes
    float*  xc     = (float*)(ws + NE * 2);      // B*L*D f32 = NE*4 bytes
    __bf16* wqb    = (__bf16*)(ws + NE * 6);
    __bf16* wkb    = (__bf16*)(ws + NE * 6 + WE * 2);
    __bf16* wvb    = (__bf16*)(ws + NE * 6 + WE * 4);
    __bf16* wob    = (__bf16*)(ws + NE * 6 + WE * 6);
    __bf16* qbuf   = (__bf16*)(ws + NE * 6 + WE * 8);
    __bf16* kbuf   = (__bf16*)(ws + NE * 8 + WE * 8);
    __bf16* vtb    = (__bf16*)(ws + NE * 10 + WE * 8);   // [B, D, L] transposed v
    __bf16* attn   = (__bf16*)(ws + NE * 12 + WE * 8);   // total 64 MiB

    // 1) fp32 -> bf16 converts
    cvt_f32_bf16<<<(unsigned)((NE + 255) / 256), 256, 0, stream>>>(Query, xq, NE);
    cvt_f32_bf16<<<(unsigned)((NE + 255) / 256), 256, 0, stream>>>(Key,   xk, NE);
    cvt_f32_bf16<<<(unsigned)((NE + 255) / 256), 256, 0, stream>>>(Value, xv, NE);
    cvt_f32_bf16<<<(unsigned)((WE + 255) / 256), 256, 0, stream>>>(Wq, wqb, WE);
    cvt_f32_bf16<<<(unsigned)((WE + 255) / 256), 256, 0, stream>>>(Wk, wkb, WE);
    cvt_f32_bf16<<<(unsigned)((WE + 255) / 256), 256, 0, stream>>>(Wv, wvb, WE);
    cvt_f32_bf16<<<(unsigned)((WE + 255) / 256), 256, 0, stream>>>(Wo, wob, WE);

    // 2) projections (NT): M = B*L = 4096, N = D, K = D; 128x256 block tiles
    dim3 gproj((B_ * L_) / 128, D_ / 256, 1), blk(256);
    gemm_nt<<<gproj, blk, 0, stream>>>(xq, D_, 0, wqb, D_, 0, bq, D_,
                                       nullptr, qbuf, 0, 0, D_);
    gemm_nt<<<gproj, blk, 0, stream>>>(xk, D_, 0, wkb, D_, 0, bk, D_,
                                       nullptr, kbuf, 0, 0, D_);
    gemm_nt<<<gproj, blk, 0, stream>>>(xv, D_, 0, wvb, D_, 0, bv, D_,
                                       nullptr, vtb, 0, 1, D_);   // transposed store

    // 3) head-shared mixing matrix
    common_kernel<<<B_ * L_, 256, 0, stream>>>(adj, imf1, imf2, imf3, imf4, imf5,
                                               lambdas, mask, common);

    // 4) xc[b] = common[b] @ v[b]  (batched NT, f32 out)
    dim3 gxc(L_ / 128, D_ / 256, B_);
    gemm_nt<<<gxc, blk, 0, stream>>>(common, L_, (size_t)L_ * L_,
                                     vtb, L_, (size_t)D_ * L_,
                                     nullptr, L_, xc, nullptr,
                                     (size_t)L_ * D_, 2, D_);

    // 5) fused attention + combine
    attn_kernel<<<dim3(L_ / 16, H_, B_), dim3(128), 0, stream>>>(
        qbuf, kbuf, vtb, xc, lambdas, mask, attn);

    // 6) output projection -> f32 d_out
    gemm_nt<<<gproj, blk, 0, stream>>>(attn, D_, 0, wob, D_, 0, bo, D_,
                                       out, nullptr, 0, 2, D_);
}
